// SplitSynthesisLayer_43980465111654
// MI455X (gfx1250) — matrix operand order repaired and checked by reading, not compiled, not run on hardware
//
#include <hip/hip_runtime.h>
#include <hip/hip_bf16.h>

// ---------------- config (matches reference) ----------------
#define RES     512
#define NB      16          // batch
#define CIN     32
#define COUT    32
#define LRELU   0.2f
#define SQRT2   1.4142135623730951f
#define AFFG    0.044194173824159216f   // 1/sqrt(512)
#define CLAMP   256.0f

typedef __bf16 bf16_t;
typedef __attribute__((ext_vector_type(16))) __bf16 v16bf;
typedef __attribute__((ext_vector_type(8)))  __bf16 v8bf;
typedef __attribute__((ext_vector_type(8)))  float  v8f;
typedef __attribute__((ext_vector_type(4)))  unsigned int u32x4;
typedef __attribute__((ext_vector_type(8)))  int i32x8;
typedef __attribute__((ext_vector_type(4)))  int i32x4;

// A matrices, fragment-swizzled per sample:
//   frag f = (phase*5 + chunk)*2 + mtile   (20 frags)
//   frag block = [half(2)][lane(32)][8 bf16]  = 1024 B
// Per-sample total = 20 * 1024 B = 20480 B  (10240 bf16)
#define A_ELEMS (COUT*320)         // 10240
#define A_BYTES (A_ELEMS*2)        // 20480

// conv tile
#define TILE_W  128
#define TILE_H  4
#define XROWS   6                  // TILE_H + 2 halo rows
#define XCOLS   136                // col 0 <-> gx = x0-4 (keeps float4 loads 16B aligned)
#define CHS     40                 // channel-innermost stride (32 used + 8 pad): 80B, 16B-aligned,
                                   // 20-bank lane stride -> conflict-free b128 gathers
#define X_ELEMS (XROWS*XCOLS*CHS)  // 32640
#define X_BYTES (X_ELEMS*2)        // 65280
#define LDS_BYTES (A_BYTES + X_BYTES)   // 85760 bytes dynamic LDS (WGP has 320KB)

// -------------------------------------------------------------
// Stage 1: styles -> modulate -> demod -> fragment-swizzled bf16 A
// K (within phase) = tap*16 + c', taps padded 9->10 (tap 9 = zeros)
// -------------------------------------------------------------
__global__ __launch_bounds__(256) void prep_weights_kernel(
    const float* __restrict__ ws0, const float* __restrict__ ws1,
    const float* __restrict__ weight, const float* __restrict__ aff_w0,
    const float* __restrict__ aff_b0, const float* __restrict__ aff_w1,
    const float* __restrict__ aff_b1, bf16_t* __restrict__ wsA)
{
    __shared__ float styles[32];
    __shared__ float dmod[COUT][2];
    const int b = blockIdx.x;
    const int tid = threadIdx.x;

    if (tid < 32) {
        const int j = tid & 15;
        const float* wv = (tid < 16) ? (ws0 + b * 512) : (ws1 + b * 512);
        const float* aw = (tid < 16) ? (aff_w0 + j * 512) : (aff_w1 + j * 512);
        const float  ab = (tid < 16) ? aff_b0[j] : aff_b1[j];
        float acc = 0.f;
        for (int d = 0; d < 512; ++d) acc += wv[d] * aw[d];
        float y = acc * AFFG + ab;
        y = (y > 0.f ? y : LRELU * y) * SQRT2;
        styles[tid] = y;
    }
    __syncthreads();

    if (tid < COUT) {
        const int o = tid;
        #pragma unroll
        for (int p = 0; p < 2; ++p) {
            float s = 0.f;
            for (int c = 0; c < 16; ++c) {
                const float st = styles[p * 16 + c];
                #pragma unroll
                for (int t = 0; t < 9; ++t) {
                    const float w = weight[(o * CIN + p * 16 + c) * 9 + t] * st;
                    s += w * w;
                }
            }
            dmod[o][p] = rsqrtf(s + 1e-8f);
        }
    }
    __syncthreads();

    // fragment-swizzled write: idx = ((f*2 + half)*32 + l)*8 + (e & 7), half = e >> 3
    for (int idx = tid; idx < A_ELEMS; idx += 256) {
        const int e  = ((idx >> 8) & 1) * 8 + (idx & 7);
        const int l  = (idx >> 3) & 31;
        const int f  = idx >> 9;            // 0..19
        const int mt = f & 1;
        const int pq = f >> 1;              // 0..9
        const int p  = pq / 5, q = pq % 5;
        const int n  = l & 15, hi = l >> 4;
        const int o  = mt * 16 + n;
        const int kk = 32 * q + 16 * hi + e;  // K within phase
        const int t  = kk >> 4, c = kk & 15;
        float v = 0.f;
        if (t < 9)
            v = weight[(o * CIN + p * 16 + c) * 9 + t] * styles[p * 16 + c] * dmod[o][p];
        wsA[(size_t)b * A_ELEMS + idx] = (bf16_t)v;
    }
}

// tap offset (elements) into channel-innermost x tile; pad tap 9 clamps to a
// valid tap (its A entries are exact zeros, so the product is 0, never NaN).
__device__ __forceinline__ constexpr int toff(int t)
{
    if (t > 8) t = 8;
    return ((t / 3) * XCOLS + (t % 3)) * CHS;
}

// -------------------------------------------------------------
// Stage 2: fused dual-output implicit-GEMM conv via BF16 WMMA
// -------------------------------------------------------------
extern __shared__ char smem[];

__global__ __launch_bounds__(256) void conv_wmma_kernel(
    const float* __restrict__ x, const bf16_t* __restrict__ wsA,
    const float* __restrict__ bias, const float* __restrict__ noise_const,
    const float* __restrict__ noise_strength, float* __restrict__ out)
{
    bf16_t* As = (bf16_t*)smem;                 // fragment-swizzled A (20 KB), offset 0
    bf16_t* xs = (bf16_t*)(smem + A_BYTES);     // [row][col][ch] bf16, ch stride CHS

    const int tid = threadIdx.x;
    const int lane = tid & 31;
    const int wave = tid >> 5;
    const int b  = blockIdx.z;
    const int y0 = blockIdx.y * TILE_H;
    const int x0 = blockIdx.x * TILE_W;

    // ---- stage A (20480 contiguous bytes) into LDS via the Tensor Data Mover ----
#if __has_builtin(__builtin_amdgcn_tensor_load_to_lds) && __has_builtin(__builtin_amdgcn_s_wait_tensorcnt)
    if (wave == 0) {
        const unsigned long long ga = (unsigned long long)(wsA + (size_t)b * A_ELEMS);
        const unsigned ldsoff = (unsigned)(unsigned long long)As;   // low 32 bits = LDS byte offset
        u32x4 g0;
        g0[0] = 1u;                                   // count=1, user descriptor
        g0[1] = ldsoff;                               // lds_addr
        g0[2] = (unsigned)ga;                         // global_addr[31:0]
        g0[3] = ((unsigned)(ga >> 32) & 0x01ffffffu)  // global_addr[56:32]
                | 0x80000000u;                        // type=2 ("image")
        i32x8 g1;
        g1[0] = 1 << 16;                              // data_size=1 (2 bytes), mask=0
        g1[1] = (int)((A_ELEMS & 0xffff) << 16);      // tensor_dim0[15:0]
        g1[2] = (int)(A_ELEMS >> 16);                 // tensor_dim0[31:16], tensor_dim1=0
        g1[3] = (int)((A_ELEMS & 0xffff) << 16);      // tile_dim0 = 10240
        g1[4] = 0;                                    // tile_dim1/2 = 0 (1-D tile)
        g1[5] = A_ELEMS;                              // tensor_dim0_stride (low 32)
        g1[6] = 0;
        g1[7] = 0;
        const i32x4 gz4 = {0, 0, 0, 0};               // groups 2/3 unused
        const i32x8 gz8 = {0, 0, 0, 0, 0, 0, 0, 0};
        __builtin_amdgcn_tensor_load_to_lds(g0, g1, gz4, gz4, gz8, 0);
        __builtin_amdgcn_s_wait_tensorcnt(0);
    }
#else
    {
        const uint4* src = (const uint4*)(wsA + (size_t)b * A_ELEMS);
        uint4* dst = (uint4*)As;
        #pragma unroll
        for (int i = tid; i < A_ELEMS / 8; i += 256) dst[i] = src[i];
    }
#endif
    // ---- stage x tile (f32 -> bf16, transpose to channel-innermost) ----
    {
        const int NCOLG = XCOLS / 4;             // 34
        const int NG = 8 * XROWS * NCOLG;        // 8 chan-groups * 6 rows * 34 = 1632
        for (int i = tid; i < NG; i += 256) {
            const int colg = i % NCOLG;
            const int rem  = i / NCOLG;
            const int r    = rem % XROWS;
            const int cg   = rem / XROWS;        // 0..7 -> channels 4cg..4cg+3
            const int gy = y0 - 1 + r;
            const int gx = x0 - 4 + colg * 4;
            const bool inb = (gy >= 0 && gy < RES && gx >= 0 && gx + 3 < RES);
            float f[16];                         // [ch][col]
            #pragma unroll
            for (int u = 0; u < 4; ++u) {
                float4 v = {0.f, 0.f, 0.f, 0.f};
                if (inb)
                    v = *(const float4*)(x +
                        (((size_t)(b * CIN + cg * 4 + u) * RES + gy) * RES + gx));
                f[u * 4 + 0] = v.x; f[u * 4 + 1] = v.y;
                f[u * 4 + 2] = v.z; f[u * 4 + 3] = v.w;
            }
            #pragma unroll
            for (int u = 0; u < 4; ++u) {        // column u: 4 contiguous channels
                union { bf16_t h[4]; uint2 qw; } pk;
                pk.h[0] = (bf16_t)f[0 * 4 + u];
                pk.h[1] = (bf16_t)f[1 * 4 + u];
                pk.h[2] = (bf16_t)f[2 * 4 + u];
                pk.h[3] = (bf16_t)f[3 * 4 + u];
                *(uint2*)&xs[(r * XCOLS + colg * 4 + u) * CHS + cg * 4] = pk.qw;
            }
        }
    }
    __syncthreads();

    const int wr   = wave >> 1;          // tile row 0..3
    const int xh   = (wave & 1) * 64;    // column half
    const int n    = lane & 15;          // WMMA column (B) / M row (A)
    const int hi   = lane >> 4;          // lane half: K sub-range / tap parity
    const int y    = y0 + wr;
    const float nstr = noise_strength[0];
    const size_t humanOff = (size_t)NB * COUT * RES * RES;
    const bf16_t* afb = As + lane * 8;   // per-lane A base; frags at immediate offsets

    for (int it = 0; it < 4; ++it) {
        const int px0 = xh + it * 16;
        const int gx  = x0 + px0 + n;
        const float nv = noise_const[y * RES + gx] * nstr;
        const int pixbase = (wr * XCOLS + px0 + n + 3) * CHS;
        const size_t chanpix = ((size_t)(b * COUT) * RES + y) * RES + gx;

        v8f acc0 = {};   // output channels 0..15
        v8f acc1 = {};   // output channels 16..31

        #pragma unroll
        for (int p = 0; p < 2; ++p) {
            #pragma unroll
            for (int q = 0; q < 5; ++q) {
                // ---- B fragment: 16 contiguous channels of tap (2q + hi) ----
                const int off = pixbase + p * 16 +
                                (hi ? toff(2 * q + 1) : toff(2 * q));
                const v8bf bl = *(const v8bf*)(xs + off);
                const v8bf bh = *(const v8bf*)(xs + off + 8);
                v16bf bfrag;
                #pragma unroll
                for (int e = 0; e < 8; ++e) { bfrag[e] = bl[e]; bfrag[8 + e] = bh[e]; }

                // ---- A fragments: base VGPR + immediate offsets, 16B lane stride ----
                const int f0 = ((p * 5 + q) * 2 + 0) * 512;   // elements
                const int f1 = ((p * 5 + q) * 2 + 1) * 512;
                const v8bf a0l = *(const v8bf*)(afb + f0);
                const v8bf a0h = *(const v8bf*)(afb + f0 + 256);
                const v8bf a1l = *(const v8bf*)(afb + f1);
                const v8bf a1h = *(const v8bf*)(afb + f1 + 256);
                v16bf a0, a1;
                #pragma unroll
                for (int e = 0; e < 8; ++e) {
                    a0[e] = a0l[e]; a0[8 + e] = a0h[e];
                    a1[e] = a1l[e]; a1[8 + e] = a1h[e];
                }
                acc0 = __builtin_amdgcn_wmma_f32_16x16x32_bf16(
                           false, a0, false, bfrag, (short)0, acc0, false, false);
                acc1 = __builtin_amdgcn_wmma_f32_16x16x32_bf16(
                           false, a1, false, bfrag, (short)0, acc1, false, false);
            }
            // ---- epilogue: p==0 snapshot = face, p==1 = human ----
            float* dst = out + (p ? humanOff : (size_t)0);
            #pragma unroll
            for (int mt = 0; mt < 2; ++mt) {
                const v8f acc = mt ? acc1 : acc0;
                const int obase = mt * 16;
                #pragma unroll
                for (int r = 0; r < 8; ++r) {
                    const int o = obase + r + (hi ? 8 : 0);
                    const float bv = hi ? bias[obase + 8 + r] : bias[obase + r];
                    float v = acc[r] + nv + bv;
                    v = (v > 0.f ? v : LRELU * v) * SQRT2;
                    v = fminf(fmaxf(v, -CLAMP), CLAMP);
                    dst[chanpix + (size_t)o * (RES * RES)] = v;
                }
            }
        }
    }
}

// -------------------------------------------------------------
extern "C" void kernel_launch(void* const* d_in, const int* in_sizes, int n_in,
                              void* d_out, int out_size, void* d_ws, size_t ws_size,
                              hipStream_t stream)
{
    (void)in_sizes; (void)n_in; (void)out_size; (void)ws_size;
    const float* x      = (const float*)d_in[0];
    const float* ws0    = (const float*)d_in[1];
    const float* ws1    = (const float*)d_in[2];
    const float* weight = (const float*)d_in[3];
    const float* bias   = (const float*)d_in[4];
    const float* aw0    = (const float*)d_in[5];
    const float* ab0    = (const float*)d_in[6];
    const float* aw1    = (const float*)d_in[7];
    const float* ab1    = (const float*)d_in[8];
    const float* ncst   = (const float*)d_in[9];
    const float* nstr   = (const float*)d_in[10];
    bf16_t* wsA = (bf16_t*)d_ws;
    float*  out = (float*)d_out;

    prep_weights_kernel<<<NB, 256, 0, stream>>>(ws0, ws1, weight, aw0, ab0, aw1, ab1, wsA);

    dim3 grid(RES / TILE_W, RES / TILE_H, NB);   // (4, 128, 16)
    conv_wmma_kernel<<<grid, 256, LDS_BYTES, stream>>>(x, wsA, bias, ncst, nstr, out);
}
